// Attention_38534446580369
// MI455X (gfx1250) — matrix operand order repaired and checked by reading, not compiled, not run on hardware
//
#include <hip/hip_runtime.h>
#include <hip/hip_bf16.h>
#include <math.h>

// ---------------------------------------------------------------------------
// Causal attention with fused QKV projection for gfx1250 (MI455X).
// Kernel 1: QKV GEMM (f32 in -> f16 out, 1/sqrt(d) folded into q), WMMA f16.
// Kernel 2: flash attention, online softmax, WMMA f16, f32 out.
// Round 6: sched_barrier(0) between batched fragment loads and WMMA groups
//          so ds_load latency overlaps across WMMAs instead of 8x wait(0).
// ---------------------------------------------------------------------------

typedef __attribute__((ext_vector_type(16))) _Float16 v16h;
typedef __attribute__((ext_vector_type(8)))  _Float16 v8h;
typedef __attribute__((ext_vector_type(4)))  _Float16 v4h;
typedef __attribute__((ext_vector_type(8)))  float    v8f;
typedef int vi4 __attribute__((vector_size(16)));

#define D_MODEL 1024
#define DHEAD   256
#define TSEQ    512
#define NBATCH  64

#define AS1 __attribute__((address_space(1)))
#define AS3 __attribute__((address_space(3)))

#if defined(__HIP_DEVICE_COMPILE__) && __has_builtin(__builtin_amdgcn_global_load_async_to_lds_b128)
#define HAVE_ASYNC_LDS 1
#else
#define HAVE_ASYNC_LDS 0
#endif

union HFrag { v16h v; v8h h[2]; };

__device__ __forceinline__ void sched_fence() {
#if defined(__HIP_DEVICE_COMPILE__) && __has_builtin(__builtin_amdgcn_sched_barrier)
  __builtin_amdgcn_sched_barrier(0);   // no instruction may cross this point
#endif
}

// A-fragment (16x32 f16, M x K), ISA 7.12.2 layout.
__device__ __forceinline__ v16h load_a_frag(const _Float16* row0, int pitch_h, int kbase) {
  const int lane = threadIdx.x & 31;
  const int r = lane & 15, h = lane >> 4;
  const _Float16* p = row0 + r * pitch_h + kbase + h * 8;
  HFrag f;
  f.h[0] = *(const v8h*)(p);
  f.h[1] = *(const v8h*)(p + 16);
  return f.v;
}

// B-fragment (32x16 f16, K x N): lane = N column (lane%16), elements over K in table order.
__device__ __forceinline__ v16h load_b_frag(const _Float16* base, int pitch_h, int n0) {
  const int lane = threadIdx.x & 31;
  const int h = lane >> 4;
  const _Float16* p = base + (h * 8) * pitch_h + n0 + (lane & 15);
  HFrag f;
#pragma unroll
  for (int s = 0; s < 8; ++s) {
    f.v[s]     = p[s * pitch_h];
    f.v[s + 8] = p[(16 + s) * pitch_h];
  }
  return f.v;
}

__device__ __forceinline__ v8f wmma_f16(v16h a, v16h b, v8f c) {
  return __builtin_amdgcn_wmma_f32_16x16x32_f16(false, a, false, b, (short)0, c,
                                                false, false);
}

__device__ __forceinline__ v4h cvt4(float4 f) {
  v4h h;
  h.x = (_Float16)f.x; h.y = (_Float16)f.y;
  h.z = (_Float16)f.z; h.w = (_Float16)f.w;
  return h;
}

#if HAVE_ASYNC_LDS
// 16-byte async DMA global -> LDS, tracked with ASYNCcnt.
__device__ __forceinline__ void async_b128(const _Float16* g, _Float16* l) {
  __builtin_amdgcn_global_load_async_to_lds_b128(
      (vi4*)(void*)g, (AS3 vi4*)(unsigned long long)(size_t)l, 0, 0);
}
__device__ __forceinline__ void wait_async0() {
#if __has_builtin(__builtin_amdgcn_s_wait_asynccnt)
  __builtin_amdgcn_s_wait_asynccnt(0);
#else
  asm volatile("s_wait_asynccnt 0x0" ::: "memory");
#endif
}
#endif

// ---------------------------------------------------------------------------
// Kernel 1: [32768,1024] x [1024,256] + bias -> f16   (grid.y selects q/k/v)
// Block 256 thr (8 waves). Tile 64(M) x 256(N). Waves: 4(M) x 2(N-half).
// ---------------------------------------------------------------------------
__global__ __launch_bounds__(256) void qkv_gemm_kernel(
    const float* __restrict__ x,
    const float* __restrict__ Wq, const float* __restrict__ bq,
    const float* __restrict__ Wk, const float* __restrict__ bk,
    const float* __restrict__ Wv, const float* __restrict__ bv,
    _Float16* __restrict__ qh, _Float16* __restrict__ kh,
    _Float16* __restrict__ vh)
{
  __shared__ alignas(16) _Float16 Xs[64 * 40];
  __shared__ alignas(16) _Float16 Ws[32 * 264];

  const int mb = blockIdx.x;
  const int nb = blockIdx.y;
  const float* W    = (nb == 0) ? Wq : (nb == 1) ? Wk : Wv;
  const float* bias = (nb == 0) ? bq : (nb == 1) ? bk : bv;
  _Float16*    out  = (nb == 0) ? qh : (nb == 1) ? kh : vh;
  const float  scale = (nb == 0) ? 0.0625f : 1.0f;

  const int row0 = mb * 64;
  const int tid  = threadIdx.x;
  const int lane = tid & 31;
  const int wave = tid >> 5;
  const int wm   = wave >> 1;
  const int wn   = wave & 1;

  const int xr0  = tid >> 3;           // x chunk: 2 float4 per thread
  const int xc0  = (tid & 7) * 4;
  const int wr0  = tid >> 6;           // W chunk: 8 float4 per thread
  const int wc0  = (tid & 63) * 4;

  float4 xreg[2], wreg[8];
  {
    const float* xp = x + (size_t)row0 * D_MODEL;
#pragma unroll
    for (int j = 0; j < 2; ++j)
      xreg[j] = *(const float4*)(xp + (size_t)(xr0 + j * 32) * D_MODEL + xc0);
#pragma unroll
    for (int j = 0; j < 8; ++j)
      wreg[j] = *(const float4*)(W + (size_t)(wr0 + j * 4) * DHEAD + wc0);
  }

  v8f acc[8] = {};

  for (int kc = 0; kc < D_MODEL; kc += 32) {
    __syncthreads();
#pragma unroll
    for (int j = 0; j < 2; ++j)
      *(v4h*)(&Xs[(xr0 + j * 32) * 40 + xc0]) = cvt4(xreg[j]);
#pragma unroll
    for (int j = 0; j < 8; ++j)
      *(v4h*)(&Ws[(wr0 + j * 4) * 264 + wc0]) = cvt4(wreg[j]);
    __syncthreads();

    const int kn = kc + 32;
    if (kn < D_MODEL) {
      const float* xp = x + (size_t)row0 * D_MODEL + kn;
#pragma unroll
      for (int j = 0; j < 2; ++j)
        xreg[j] = *(const float4*)(xp + (size_t)(xr0 + j * 32) * D_MODEL + xc0);
      const float* wp = W + (size_t)kn * DHEAD;
#pragma unroll
      for (int j = 0; j < 8; ++j)
        wreg[j] = *(const float4*)(wp + (size_t)(wr0 + j * 4) * DHEAD + wc0);
    }

    // Batch all fragment loads; the sched barrier pins them above the WMMA
    // group so all A (b128) and B (16-bit) fragment ds_loads stay in flight with staged dscnt waits.
    const v16h a = load_a_frag(Xs + wm * 16 * 40, 40, 0);
    v16h bf[8];
#pragma unroll
    for (int nt = 0; nt < 8; ++nt)
      bf[nt] = load_b_frag(Ws, 264, wn * 128 + nt * 16);
    sched_fence();
#pragma unroll
    for (int nt = 0; nt < 8; ++nt)
      acc[nt] = wmma_f16(a, bf[nt], acc[nt]);
  }

  const int half = lane >> 4, n = lane & 15;
#pragma unroll
  for (int nt = 0; nt < 8; ++nt) {
    const int col = wn * 128 + nt * 16 + n;
    const float bcol = bias[col];
#pragma unroll
    for (int r = 0; r < 8; ++r) {
      const int row = row0 + wm * 16 + r + half * 8;
      out[(size_t)row * DHEAD + col] = (_Float16)((acc[nt][r] + bcol) * scale);
    }
  }
}

// ---------------------------------------------------------------------------
// Kernel 2: causal flash attention on f16 q/k/v, f32 out.
// Block 256 thr (8 waves): one batch x 64 q-rows. Waves wm 0..3 x wd 0..1.
// V tiles double-buffered + async global->LDS DMA; K transposed via regs.
// ---------------------------------------------------------------------------
__global__ __launch_bounds__(256) void attn_kernel(
    const _Float16* __restrict__ qh, const _Float16* __restrict__ kh,
    const _Float16* __restrict__ vh, float* __restrict__ out)
{
  __shared__ alignas(16) _Float16 Qs[64 * 264];
  __shared__ alignas(16) _Float16 Kt[256 * 40];
#if HAVE_ASYNC_LDS
  __shared__ alignas(16) _Float16 Vs[2][32 * 264];
#else
  __shared__ alignas(16) _Float16 Vs1[32 * 264];
#endif
  __shared__ alignas(16) _Float16 Ps[8 * 16 * 40];

  const int b  = blockIdx.y;
  const int q0 = blockIdx.x * 64;
  const size_t base = (size_t)b * TSEQ * DHEAD;

  const int tid  = threadIdx.x;
  const int lane = tid & 31;
  const int wave = tid >> 5;
  const int wm   = wave >> 1;
  const int wd   = wave & 1;
  const int half = lane >> 4, nn = lane & 15;

  const int skey = tid >> 3;           // K/V staging: key row
  const int sc0  = (tid & 7) * 32;     // 32-d span
  const int ntiles = (q0 + 64) / 32;

  // --- stage Q tile ---
  {
    const int qr = tid >> 2, qc = (tid & 3) * 64;
    const _Float16* gq = qh + base + (size_t)(q0 + qr) * DHEAD + qc;
#pragma unroll
    for (int j = 0; j < 8; ++j) {
#if HAVE_ASYNC_LDS
      async_b128(gq + j * 8, &Qs[qr * 264 + qc + j * 8]);
#else
      *(v8h*)(&Qs[qr * 264 + qc + j * 8]) = *(const v8h*)(gq + j * 8);
#endif
    }
  }

  v8h kreg[4];
#if !HAVE_ASYNC_LDS
  v8h vreg[4];
#endif

  // prefetch tile 0
  {
    const _Float16* kp = kh + base + (size_t)skey * DHEAD + sc0;
#pragma unroll
    for (int j = 0; j < 4; ++j) kreg[j] = *(const v8h*)(kp + j * 8);
#if HAVE_ASYNC_LDS
    const _Float16* vp = vh + base + (size_t)skey * DHEAD + sc0;
#pragma unroll
    for (int j = 0; j < 4; ++j)
      async_b128(vp + j * 8, &Vs[0][skey * 264 + sc0 + j * 8]);
#else
    const _Float16* vp = vh + base + (size_t)skey * DHEAD + sc0;
#pragma unroll
    for (int j = 0; j < 4; ++j) vreg[j] = *(const v8h*)(vp + j * 8);
#endif
  }

  v8f acc[8] = {};
  float m_i[8], l_i[8];
#pragma unroll
  for (int r = 0; r < 8; ++r) { m_i[r] = -INFINITY; l_i[r] = 0.0f; }

  const int row_min = q0 + wm * 16;
  const int row_max = row_min + 15;

  for (int j = 0; j < ntiles; ++j) {
    const int kb = j * 32;
#if HAVE_ASYNC_LDS
    wait_async0();                     // V tile j (and Q on j==0) landed
#endif
    __syncthreads();                   // + all waves done with previous tile
#pragma unroll
    for (int q = 0; q < 4; ++q) {
#if !HAVE_ASYNC_LDS
      *(v8h*)(&Vs1[skey * 264 + sc0 + q * 8]) = vreg[q];
#endif
#pragma unroll
      for (int e = 0; e < 8; ++e)
        Kt[(sc0 + q * 8 + e) * 40 + skey] = kreg[q][e];
    }
    __syncthreads();

    // prefetch next tile while compute runs
    if (j + 1 < ntiles) {
      const size_t g = base + (size_t)(kb + 32 + skey) * DHEAD + sc0;
#pragma unroll
      for (int q = 0; q < 4; ++q) kreg[q] = *(const v8h*)(kh + g + q * 8);
#if HAVE_ASYNC_LDS
#pragma unroll
      for (int q = 0; q < 4; ++q)
        async_b128(vh + g + q * 8, &Vs[(j + 1) & 1][skey * 264 + sc0 + q * 8]);
#else
#pragma unroll
      for (int q = 0; q < 4; ++q) vreg[q] = *(const v8h*)(vh + g + q * 8);
#endif
    }

#if HAVE_ASYNC_LDS
    const _Float16* Vcur = Vs[j & 1];
#else
    const _Float16* Vcur = Vs1;
#endif

    if (kb <= row_max) {               // skip tiles fully above the diagonal
      // --- S = q k^T (16x32), f32 accum over d=256 ---
      // Two-slot ping-pong, sched-fenced so each slot's loads stay hoisted
      // above the other slot's WMMAs.
      v8f s0 = {}, s1 = {};
      const _Float16* Qrow = Qs + wm * 16 * 264;
      v16h aA  = load_a_frag(Qrow, 264, 0);
      v16h b0A = load_b_frag(Kt, 40, 0);
      v16h b1A = load_b_frag(Kt, 40, 16);
#pragma unroll
      for (int dc = 0; dc < 256; dc += 64) {
        const int d2 = dc + 32;
        v16h aB  = load_a_frag(Qrow, 264, d2);
        v16h b0B = load_b_frag(Kt + d2 * 40, 40, 0);
        v16h b1B = load_b_frag(Kt + d2 * 40, 40, 16);
        sched_fence();
        s0 = wmma_f16(aA, b0A, s0);
        s1 = wmma_f16(aA, b1A, s1);
        if (dc + 64 < 256) {
          const int d3 = dc + 64;
          aA  = load_a_frag(Qrow, 264, d3);
          b0A = load_b_frag(Kt + d3 * 40, 40, 0);
          b1A = load_b_frag(Kt + d3 * 40, 40, 16);
        }
        sched_fence();
        s0 = wmma_f16(aB, b0B, s0);
        s1 = wmma_f16(aB, b1B, s1);
      }

      // --- causal mask on diagonal tiles ---
      if (kb + 31 > row_min) {
        const int key0 = kb + nn, key1 = kb + 16 + nn;
#pragma unroll
        for (int r = 0; r < 8; ++r) {
          const int row = row_min + r + half * 8;
          if (key0 > row) s0[r] = -INFINITY;
          if (key1 > row) s1[r] = -INFINITY;
        }
      }

      // --- online softmax (row = 16 lanes of a half-wave) ---
#pragma unroll
      for (int r = 0; r < 8; ++r) {
        float mx = fmaxf(s0[r], s1[r]);
#pragma unroll
        for (int off = 8; off >= 1; off >>= 1)
          mx = fmaxf(mx, __shfl_xor(mx, off));
        const float mnew = fmaxf(m_i[r], mx);
        const float corr = __expf(m_i[r] - mnew);
        m_i[r] = mnew;
        const float p0 = __expf(s0[r] - mnew);
        const float p1 = __expf(s1[r] - mnew);
        s0[r] = p0; s1[r] = p1;
        float rs = p0 + p1;
#pragma unroll
        for (int off = 8; off >= 1; off >>= 1)
          rs += __shfl_xor(rs, off);
        l_i[r] = l_i[r] * corr + rs;
#pragma unroll
        for (int nt = 0; nt < 8; ++nt) acc[nt][r] *= corr;
      }

      // --- P: C-layout -> A-layout via wave-private LDS ---
      _Float16* Pw = Ps + wave * 16 * 40;
#pragma unroll
      for (int r = 0; r < 8; ++r) {
        Pw[(r + half * 8) * 40 + nn]      = (_Float16)s0[r];
        Pw[(r + half * 8) * 40 + 16 + nn] = (_Float16)s1[r];
      }
      const v16h pa = load_a_frag(Pw, 40, 0);
      v16h vb[8];
#pragma unroll
      for (int nt = 0; nt < 8; ++nt)
        vb[nt] = load_b_frag(Vcur, 264, wd * 128 + nt * 16);
      sched_fence();
#pragma unroll
      for (int nt = 0; nt < 8; ++nt)
        acc[nt] = wmma_f16(pa, vb[nt], acc[nt]);
    }
  }

  // --- normalize and store f32 ---
#pragma unroll
  for (int r = 0; r < 8; ++r) {
    const float inv = 1.0f / l_i[r];
    const int row = row_min + r + half * 8;
#pragma unroll
    for (int nt = 0; nt < 8; ++nt) {
      const int col = wd * 128 + nt * 16 + nn;
      out[base + (size_t)row * DHEAD + col] = acc[nt][r] * inv;
    }
  }
}

// ---------------------------------------------------------------------------
extern "C" void kernel_launch(void* const* d_in, const int* in_sizes, int n_in,
                              void* d_out, int out_size, void* d_ws, size_t ws_size,
                              hipStream_t stream) {
  (void)in_sizes; (void)n_in; (void)out_size; (void)ws_size;
  const float* x  = (const float*)d_in[0];
  const float* Wq = (const float*)d_in[1];
  const float* bq = (const float*)d_in[2];
  const float* Wk = (const float*)d_in[3];
  const float* bk = (const float*)d_in[4];
  const float* Wv = (const float*)d_in[5];
  const float* bv = (const float*)d_in[6];

  const size_t BTD = (size_t)NBATCH * TSEQ * DHEAD;
  _Float16* qh = (_Float16*)d_ws;
  _Float16* kh = qh + BTD;
  _Float16* vh = kh + BTD;

  dim3 blk(256);
  dim3 g1((NBATCH * TSEQ) / 64, 3);
  qkv_gemm_kernel<<<g1, blk, 0, stream>>>(x, Wq, bq, Wk, bk, Wv, bv, qh, kh, vh);

  dim3 g2(TSEQ / 64, NBATCH);
  attn_kernel<<<g2, blk, 0, stream>>>(qh, kh, vh, (float*)d_out);
}
